// DeepSeekMoELayer_38027640439046
// MI455X (gfx1250) — matrix operand order repaired and checked by reading, not compiled, not run on hardware
//
#include <hip/hip_runtime.h>

// ---------------- Problem constants ----------------
#define DD   1024   // d_model
#define HS   2048   // shared hidden
#define HE   1024   // expert hidden
#define NE   8      // experts
#define TOPK 2
#define NB   4      // batch
#define NT   1024   // seq
#define NTOK (NB*NT)        // 4096 tokens
#define NASG (NTOK*TOPK)    // 8192 assignments

// GEMM tiling: 64(M) x 128(N) x 32(K) macro tile, 8 waves = 2(M) x 4(N),
// each wave owns a 32x32 output block: 2 A-frags x 2 B-frags -> 4 WMMAs/K-step.
#define BM 64
#define BN 128
#define BK 32
#define SA 34   // padded LDS row stride in halves (17 dwords) to spread banks

typedef unsigned int  u32;
typedef unsigned short u16;
typedef unsigned long long u64;
typedef __attribute__((ext_vector_type(16))) __bf16 bfx16;
typedef __attribute__((ext_vector_type(2)))  __bf16 bfx2;
typedef __attribute__((ext_vector_type(8)))  float  fx8;

// ---- f32 -> bf16 (RNE) ----
__device__ __forceinline__ u16 f2bf(float f) {
    bfx2 v; v.x = (__bf16)f; v.y = (__bf16)0.0f;
    return (u16)(__builtin_bit_cast(u32, v) & 0xffffu);
}
// packed pair: low half = lo, high half = hi
__device__ __forceinline__ u32 f2bf2(float lo, float hi) {
#if __has_builtin(__builtin_amdgcn_cvt_pk_bf16_f32)
    bfx2 v = __builtin_amdgcn_cvt_pk_bf16_f32(lo, hi);
    return __builtin_bit_cast(u32, v);
#else
    bfx2 v; v.x = (__bf16)lo; v.y = (__bf16)hi;   // fptrunc = RNE
    return __builtin_bit_cast(u32, v);
#endif
}

__device__ __forceinline__ float fast_rcp(float x) {
#if __has_builtin(__builtin_amdgcn_rcpf)
    return __builtin_amdgcn_rcpf(x);
#else
    return 1.0f / x;
#endif
}
__device__ __forceinline__ float silu_f(float x) {
    return x * fast_rcp(1.0f + __expf(-x));
}
__device__ __forceinline__ float gelu_f(float x) {
    return 0.5f * x * (1.0f + erff(x * 0.70710678118654752f));
}

// ---- CDNA5 async global->LDS (ASYNCcnt-tracked) ----
__device__ __forceinline__ void async_g2l_b32(const void* g, void* lds_ptr) {
    u32 loff = (u32)(size_t)lds_ptr;        // LDS aperture: addr[31:0] is the LDS offset
    u64 ga   = (u64)(size_t)g;
    asm volatile("global_load_async_to_lds_b32 %0, %1, off"
                 :: "v"(loff), "v"(ga) : "memory");
}
__device__ __forceinline__ void wait_async0() {
    asm volatile("s_wait_asynccnt 0" ::: "memory");
}

// ---- WMMA fragment loads (wave32), contiguous k order: element s of lane half h is k = 16h + s, A and B alike ----
__device__ __forceinline__ bfx16 load_afrag(const u16* At, int rbase, int lane) {
    int half = lane >> 4, m = lane & 15;
    const u32* p = (const u32*)(At + (size_t)(rbase + m) * SA);
    int kb2 = half ? 8 : 0;
    union { bfx16 v; u32 u[8]; } f;
#pragma unroll
    for (int j = 0; j < 8; ++j) f.u[j] = p[kb2 + j];
    return f.v;
}
__device__ __forceinline__ bfx16 load_bfrag(const u16* Bt, int nbase, int lane) {
    int half = lane >> 4, n = lane & 15;
    const u32* p = (const u32*)(Bt + (size_t)(nbase + n) * SA);
    int kb2 = half ? 8 : 0;
    union { bfx16 v; u32 u[8]; } f;
#pragma unroll
    for (int j = 0; j < 8; ++j) f.u[j] = p[kb2 + j];
    return f.v;
}
__device__ __forceinline__ fx8 wmma_bf16(bfx16 a, bfx16 b, fx8 c) {
    return __builtin_amdgcn_wmma_f32_16x16x32_bf16(false, a, false, b, (short)0, c, false, false);
}

// ================= Router =================
__global__ __launch_bounds__(256) void moe_router(
    const float* __restrict__ x, const float* __restrict__ temb,
    const float* __restrict__ Wr, const float* __restrict__ rbias,
    int* __restrict__ counts, int* __restrict__ list, float* __restrict__ gateA)
{
    int t   = blockIdx.x;
    int tid = threadIdx.x;
    int b   = t >> 10;             // NT = 1024
    __shared__ float red[256 * NE];

    float acc[NE];
#pragma unroll
    for (int e = 0; e < NE; ++e) acc[e] = 0.0f;

    for (int i = tid; i < 2 * DD; i += 256) {
        float v = (i < DD) ? x[(size_t)t * DD + i] : temb[(size_t)b * DD + (i - DD)];
        const float* wr = Wr + (size_t)i * NE;
#pragma unroll
        for (int e = 0; e < NE; ++e) acc[e] += v * wr[e];
    }
#pragma unroll
    for (int e = 0; e < NE; ++e) red[tid * NE + e] = acc[e];
    __syncthreads();
    for (int s = 128; s > 0; s >>= 1) {
        if (tid < s) {
#pragma unroll
            for (int e = 0; e < NE; ++e) red[tid * NE + e] += red[(tid + s) * NE + e];
        }
        __syncthreads();
    }
    if (tid == 0) {
        float s[NE], sel[NE];
#pragma unroll
        for (int e = 0; e < NE; ++e) {
            s[e]   = 1.0f / (1.0f + __expf(-red[e]));
            sel[e] = s[e] + rbias[e];
        }
        int b0 = 0; float v0 = sel[0];
        for (int e = 1; e < NE; ++e) if (sel[e] > v0) { v0 = sel[e]; b0 = e; }
        int b1 = -1; float v1 = -3.4e38f;
        for (int e = 0; e < NE; ++e) if (e != b0 && sel[e] > v1) { v1 = sel[e]; b1 = e; }
        float denom = s[b0] + s[b1];
        float g0, g1;
        if (denom > 1e-9f) { g0 = s[b0] / (denom + 1e-9f); g1 = s[b1] / (denom + 1e-9f); }
        else               { g0 = 0.5f; g1 = 0.5f; }
        int p0 = atomicAdd(&counts[b0], 1);
        int p1 = atomicAdd(&counts[b1], 1);
        list[b0 * NTOK + p0] = t * TOPK + 0;
        list[b1 * NTOK + p1] = t * TOPK + 1;
        gateA[t * TOPK + 0] = g0;
        gateA[t * TOPK + 1] = g1;
    }
}

// ======== Shared stage 1: mid = silu(x@w1) * (x@w3) (bf16) ========
__global__ __launch_bounds__(256) void moe_shared_gemm1(
    const float* __restrict__ x, const float* __restrict__ w1,
    const float* __restrict__ w3, u16* __restrict__ mid)
{
    __shared__ u16 At[BM * SA];
    __shared__ u16 B1t[BN * SA];
    __shared__ u16 B2t[BN * SA];
    int m0 = blockIdx.x * BM, n0 = blockIdx.y * BN;
    int tid = threadIdx.x, lane = tid & 31, wid = tid >> 5;
    int wm = wid & 1, wn = wid >> 1;
    fx8 c1[2][2] = {}, c2[2][2] = {};

    u32* A32  = (u32*)At;
    u32* B132 = (u32*)B1t;
    u32* B232 = (u32*)B2t;

    for (int k0 = 0; k0 < DD; k0 += BK) {
        bool pf = (k0 + BK) < DD;
#pragma unroll
        for (int j = 0; j < (BM * BK / 2) / 256; ++j) {      // 4 iters
            int idx = tid + 256 * j;
            int r = idx >> 4, kp = idx & 15;
            const float* src = x + (size_t)(m0 + r) * DD + k0 + kp * 2;
            A32[r * 17 + kp] = f2bf2(src[0], src[1]);
            if (pf) __builtin_prefetch(src + BK, 0, 1);
        }
#pragma unroll
        for (int j = 0; j < (BN * BK / 2) / 256; ++j) {      // 8 iters
            int idx = tid + 256 * j;
            int n = idx & 127, kp = idx >> 7;
            size_t g0i = (size_t)(k0 + kp * 2) * HS + (n0 + n);
            B132[n * 17 + kp] = f2bf2(w1[g0i], w1[g0i + HS]);
            B232[n * 17 + kp] = f2bf2(w3[g0i], w3[g0i + HS]);
            if (pf) {
                __builtin_prefetch(w1 + g0i + (size_t)BK * HS, 0, 1);
                __builtin_prefetch(w3 + g0i + (size_t)BK * HS, 0, 1);
            }
        }
        __syncthreads();
        bfx16 a0 = load_afrag(At, wm * 32 + 0,  lane);
        bfx16 a1 = load_afrag(At, wm * 32 + 16, lane);
#pragma unroll
        for (int ni = 0; ni < 2; ++ni) {
            bfx16 bb = load_bfrag(B1t, wn * 32 + ni * 16, lane);
            c1[0][ni] = wmma_bf16(a0, bb, c1[0][ni]);
            c1[1][ni] = wmma_bf16(a1, bb, c1[1][ni]);
            bb = load_bfrag(B2t, wn * 32 + ni * 16, lane);
            c2[0][ni] = wmma_bf16(a0, bb, c2[0][ni]);
            c2[1][ni] = wmma_bf16(a1, bb, c2[1][ni]);
        }
        __syncthreads();
    }
    int half = lane >> 4, nl = lane & 15;
#pragma unroll
    for (int mi = 0; mi < 2; ++mi) {
#pragma unroll
        for (int ni = 0; ni < 2; ++ni) {
#pragma unroll
            for (int r = 0; r < 8; ++r) {
                int row = m0 + wm * 32 + mi * 16 + r + half * 8;
                int col = n0 + wn * 32 + ni * 16 + nl;
                mid[(size_t)row * HS + col] = f2bf(silu_f(c1[mi][ni][r]) * c2[mi][ni][r]);
            }
        }
    }
}

// ======== Shared stage 2: sharedb = mid @ w2 (A tile via async LDS DMA) ========
__global__ __launch_bounds__(256) void moe_shared_gemm2(
    const u16* __restrict__ mid, const float* __restrict__ w2, float* __restrict__ sharedb)
{
    __shared__ u16 At[BM * SA];
    __shared__ u16 Bt[BN * SA];
    int m0 = blockIdx.x * BM, n0 = blockIdx.y * BN;
    int tid = threadIdx.x, lane = tid & 31, wid = tid >> 5;
    int wm = wid & 1, wn = wid >> 1;
    fx8 c[2][2] = {};
    u32* A32 = (u32*)At;
    u32* B32 = (u32*)Bt;

    for (int k0 = 0; k0 < HS; k0 += BK) {
        bool pf = (k0 + BK) < HS;
#pragma unroll
        for (int j = 0; j < (BM * BK / 2) / 256; ++j) {
            int idx = tid + 256 * j;
            int r = idx >> 4, kp = idx & 15;
            async_g2l_b32(mid + (size_t)(m0 + r) * HS + k0 + kp * 2, &A32[r * 17 + kp]);
        }
#pragma unroll
        for (int j = 0; j < (BN * BK / 2) / 256; ++j) {
            int idx = tid + 256 * j;
            int n = idx & 127, kp = idx >> 7;
            size_t g = (size_t)(k0 + kp * 2) * DD + (n0 + n);
            B32[n * 17 + kp] = f2bf2(w2[g], w2[g + DD]);
            if (pf) __builtin_prefetch(w2 + g + (size_t)BK * DD, 0, 1);
        }
        wait_async0();
        __syncthreads();
        bfx16 a0 = load_afrag(At, wm * 32 + 0,  lane);
        bfx16 a1 = load_afrag(At, wm * 32 + 16, lane);
#pragma unroll
        for (int ni = 0; ni < 2; ++ni) {
            bfx16 bb = load_bfrag(Bt, wn * 32 + ni * 16, lane);
            c[0][ni] = wmma_bf16(a0, bb, c[0][ni]);
            c[1][ni] = wmma_bf16(a1, bb, c[1][ni]);
        }
        __syncthreads();
    }
    int half = lane >> 4, nl = lane & 15;
#pragma unroll
    for (int mi = 0; mi < 2; ++mi)
#pragma unroll
        for (int ni = 0; ni < 2; ++ni)
#pragma unroll
            for (int r = 0; r < 8; ++r) {
                int row = m0 + wm * 32 + mi * 16 + r + half * 8;
                int col = n0 + wn * 32 + ni * 16 + nl;
                sharedb[(size_t)row * DD + col] = c[mi][ni][r];
            }
}

// ======== Grouped expert GEMM 1: h[a] = gelu(x[token(a)] @ W1e[e]) ========
__global__ __launch_bounds__(256) void moe_expert_gemm1(
    const float* __restrict__ x, const float* __restrict__ W1e,
    const int* __restrict__ counts, const int* __restrict__ list, u16* __restrict__ hbuf)
{
    int e = blockIdx.z;
    int cnt = counts[e];
    int mbase = blockIdx.x * BM;
    if (mbase >= cnt) return;
    int n0 = blockIdx.y * BN;

    __shared__ u16 At[BM * SA];
    __shared__ u16 Bt[BN * SA];
    __shared__ int toks[BM];
    int tid = threadIdx.x, lane = tid & 31, wid = tid >> 5;
    int wm = wid & 1, wn = wid >> 1;
    if (tid < BM) {
        int r = mbase + tid;
        toks[tid] = (r < cnt) ? list[e * NTOK + r] : -1;
    }
    __syncthreads();
    fx8 c[2][2] = {};
    u32* A32 = (u32*)At;
    u32* B32 = (u32*)Bt;

    for (int k0 = 0; k0 < DD; k0 += BK) {
        bool pf = (k0 + BK) < DD;
#pragma unroll
        for (int j = 0; j < (BM * BK / 2) / 256; ++j) {
            int idx = tid + 256 * j;
            int r = idx >> 4, kp = idx & 15;
            int a = toks[r];
            int tok = (a >= 0) ? (a >> 1) : 0;   // junk rows masked at store
            const float* src = x + (size_t)tok * DD + k0 + kp * 2;
            A32[r * 17 + kp] = f2bf2(src[0], src[1]);
            if (pf) __builtin_prefetch(src + BK, 0, 1);
        }
#pragma unroll
        for (int j = 0; j < (BN * BK / 2) / 256; ++j) {
            int idx = tid + 256 * j;
            int n = idx & 127, kp = idx >> 7;
            size_t g = ((size_t)e * DD + (k0 + kp * 2)) * HE + (n0 + n);
            B32[n * 17 + kp] = f2bf2(W1e[g], W1e[g + HE]);
            if (pf) __builtin_prefetch(W1e + g + (size_t)BK * HE, 0, 1);
        }
        __syncthreads();
        bfx16 a0 = load_afrag(At, wm * 32 + 0,  lane);
        bfx16 a1 = load_afrag(At, wm * 32 + 16, lane);
#pragma unroll
        for (int ni = 0; ni < 2; ++ni) {
            bfx16 bb = load_bfrag(Bt, wn * 32 + ni * 16, lane);
            c[0][ni] = wmma_bf16(a0, bb, c[0][ni]);
            c[1][ni] = wmma_bf16(a1, bb, c[1][ni]);
        }
        __syncthreads();
    }
    int half = lane >> 4, nl = lane & 15;
#pragma unroll
    for (int mi = 0; mi < 2; ++mi)
#pragma unroll
        for (int ni = 0; ni < 2; ++ni)
#pragma unroll
            for (int r = 0; r < 8; ++r) {
                int rl = wm * 32 + mi * 16 + r + half * 8;
                int a  = toks[rl];
                if (a >= 0) {
                    int col = n0 + wn * 32 + ni * 16 + nl;
                    hbuf[(size_t)a * HE + col] = f2bf(gelu_f(c[mi][ni][r]));
                }
            }
}

// ======== Grouped expert GEMM 2: contrib[a] = gate[a] * (h[a] @ W2e[e]) ========
// A tile (bf16 hbuf) loaded with async global->LDS DMA.
__global__ __launch_bounds__(256) void moe_expert_gemm2(
    const u16* __restrict__ hbuf, const float* __restrict__ W2e,
    const int* __restrict__ counts, const int* __restrict__ list,
    const float* __restrict__ gateA, float* __restrict__ contrib)
{
    int e = blockIdx.z;
    int cnt = counts[e];
    int mbase = blockIdx.x * BM;
    if (mbase >= cnt) return;
    int n0 = blockIdx.y * BN;

    __shared__ u16 At[BM * SA];
    __shared__ u16 Bt[BN * SA];
    __shared__ int toks[BM];
    int tid = threadIdx.x, lane = tid & 31, wid = tid >> 5;
    int wm = wid & 1, wn = wid >> 1;
    if (tid < BM) {
        int r = mbase + tid;
        toks[tid] = (r < cnt) ? list[e * NTOK + r] : -1;
    }
    __syncthreads();
    fx8 c[2][2] = {};
    u32* A32 = (u32*)At;
    u32* B32 = (u32*)Bt;

    for (int k0 = 0; k0 < HE; k0 += BK) {
        bool pf = (k0 + BK) < HE;
#pragma unroll
        for (int j = 0; j < (BM * BK / 2) / 256; ++j) {
            int idx = tid + 256 * j;
            int r = idx >> 4, kp = idx & 15;
            int a = toks[r];
            int as = (a >= 0) ? a : 0;           // junk rows masked at store
            async_g2l_b32(hbuf + (size_t)as * HE + k0 + kp * 2, &A32[r * 17 + kp]);
        }
#pragma unroll
        for (int j = 0; j < (BN * BK / 2) / 256; ++j) {
            int idx = tid + 256 * j;
            int n = idx & 127, kp = idx >> 7;
            size_t g = ((size_t)e * HE + (k0 + kp * 2)) * DD + (n0 + n);
            B32[n * 17 + kp] = f2bf2(W2e[g], W2e[g + DD]);
            if (pf) __builtin_prefetch(W2e + g + (size_t)BK * DD, 0, 1);
        }
        wait_async0();
        __syncthreads();
        bfx16 a0 = load_afrag(At, wm * 32 + 0,  lane);
        bfx16 a1 = load_afrag(At, wm * 32 + 16, lane);
#pragma unroll
        for (int ni = 0; ni < 2; ++ni) {
            bfx16 bb = load_bfrag(Bt, wn * 32 + ni * 16, lane);
            c[0][ni] = wmma_bf16(a0, bb, c[0][ni]);
            c[1][ni] = wmma_bf16(a1, bb, c[1][ni]);
        }
        __syncthreads();
    }
    int half = lane >> 4, nl = lane & 15;
#pragma unroll
    for (int mi = 0; mi < 2; ++mi)
#pragma unroll
        for (int ni = 0; ni < 2; ++ni)
#pragma unroll
            for (int r = 0; r < 8; ++r) {
                int rl = wm * 32 + mi * 16 + r + half * 8;
                int a  = toks[rl];
                if (a >= 0) {
                    int col = n0 + wn * 32 + ni * 16 + nl;
                    contrib[(size_t)a * DD + col] = gateA[a] * c[mi][ni][r];
                }
            }
}

// ======== Combine: out = (shared + contrib[2t] + contrib[2t+1]) / 3 ========
__global__ __launch_bounds__(256) void moe_combine(
    const float4* __restrict__ sharedb, const float4* __restrict__ contrib,
    float4* __restrict__ out)
{
    int i = blockIdx.x * 256 + threadIdx.x;   // over NTOK*DD/4 float4s
    int t = i >> 8;                           // DD/4 = 256 float4 per row
    int cidx = i & 255;
    size_t base = (size_t)(t << 1) * (DD / 4) + cidx;
    float4 s = sharedb[i];
    float4 a = contrib[base];
    float4 b = contrib[base + DD / 4];
    float4 o;
    o.x = (s.x + a.x + b.x) * (1.0f / 3.0f);
    o.y = (s.y + a.y + b.y) * (1.0f / 3.0f);
    o.z = (s.z + a.z + b.z) * (1.0f / 3.0f);
    o.w = (s.w + a.w + b.w) * (1.0f / 3.0f);
    out[i] = o;
}

// ================= Launch =================
extern "C" void kernel_launch(void* const* d_in, const int* in_sizes, int n_in,
                              void* d_out, int out_size, void* d_ws, size_t ws_size,
                              hipStream_t stream) {
    const float* x     = (const float*)d_in[0];
    const float* temb  = (const float*)d_in[1];
    const float* Wr    = (const float*)d_in[2];
    const float* rbias = (const float*)d_in[3];
    const float* w1    = (const float*)d_in[4];
    const float* w3    = (const float*)d_in[5];
    const float* w2    = (const float*)d_in[6];
    const float* W1e   = (const float*)d_in[7];
    const float* W2e   = (const float*)d_in[8];

    char* ws = (char*)d_ws;
    int*   counts  = (int*)(ws);
    int*   list    = (int*)(ws + 256);
    float* gateA   = (float*)(ws + 256 + (size_t)NE * NTOK * 4);
    size_t off     = 256 + (size_t)NE * NTOK * 4 + (size_t)NASG * 4;
    u16*   mid     = (u16*)(ws + off);
    float* sharedb = (float*)(ws + off + (size_t)16777216);
    u16*   hbuf    = (u16*)(ws + off + (size_t)2 * 16777216);
    float* contrib = (float*)(ws + off + (size_t)3 * 16777216);

    hipMemsetAsync(counts, 0, NE * sizeof(int), stream);

    moe_router<<<NTOK, 256, 0, stream>>>(x, temb, Wr, rbias, counts, list, gateA);
    moe_shared_gemm1<<<dim3(NTOK / BM, HS / BN), 256, 0, stream>>>(x, w1, w3, mid);
    moe_shared_gemm2<<<dim3(NTOK / BM, DD / BN), 256, 0, stream>>>(mid, w2, sharedb);
    moe_expert_gemm1<<<dim3(NTOK / BM, HE / BN, NE), 256, 0, stream>>>(x, W1e, counts, list, hbuf);
    moe_expert_gemm2<<<dim3(NTOK / BM, DD / BN, NE), 256, 0, stream>>>(hbuf, W2e, counts, list, gateA, contrib);
    moe_combine<<<(NTOK * DD / 4) / 256, 256, 0, stream>>>(
        (const float4*)sharedb, (const float4*)contrib, (float4*)d_out);
}
